// MultiHeadAttention_26156350833328
// MI455X (gfx1250) — matrix operand order repaired and checked by reading, not compiled, not run on hardware
//
#include <hip/hip_runtime.h>
#include <hip/hip_bf16.h>

#define SDIM   2048
#define DMODEL 1024
#define NHEADS 16
#define HDIM   64
#define BATCH  2
#define MROWS  (BATCH * SDIM)      // 4096
#define SCALE_F 0.125f             // 1/sqrt(64)

typedef __bf16 bf16_t;
typedef __attribute__((ext_vector_type(16))) __bf16 v16bf;
typedef __attribute__((ext_vector_type(8)))  __bf16 v8bf;
typedef __attribute__((ext_vector_type(4)))  __bf16 v4bf;
typedef __attribute__((ext_vector_type(8)))  float  v8f;
typedef __attribute__((ext_vector_type(4)))  float  v4f;

// A-fragment (16-bit A 16x32): lane<16 holds K 0..15,
// lane>=16 holds K 16..31. Caller passes p = row_base + (half*16).
static __device__ __forceinline__ v16bf load_a_frag(const bf16_t* p) {
  union { v16bf v; v8bf h[2]; } u;
  u.h[0] = *(const v8bf*)(p);
  u.h[1] = *(const v8bf*)(p + 8);
  return u.v;
}
// B-fragment (16-bit B 32x16): lane<16 holds K 0..15, lane>=16 holds K 16..31.
// Caller passes p = row_base + (half*16).
static __device__ __forceinline__ v16bf load_b_frag(const bf16_t* p) {
  union { v16bf v; v8bf h[2]; } u;
  u.h[0] = *(const v8bf*)(p);
  u.h[1] = *(const v8bf*)(p + 8);
  return u.v;
}
static __device__ __forceinline__ v8f wmma_bf16(v16bf a, v16bf b, v8f c) {
  return __builtin_amdgcn_wmma_f32_16x16x32_bf16(false, a, false, b,
                                                 (short)0, c, false, false);
}

__global__ void __launch_bounds__(256)
cvt_f32_to_bf16(const float* __restrict__ in, bf16_t* __restrict__ out, int n4) {
  int i = blockIdx.x * blockDim.x + threadIdx.x;
  if (i < n4) {
    v4f f = *(const v4f*)(in + (size_t)i * 4);
    v4bf o;
    o[0] = (bf16_t)f[0]; o[1] = (bf16_t)f[1];
    o[2] = (bf16_t)f[2]; o[3] = (bf16_t)f[3];
    *(v4bf*)(out + (size_t)i * 4) = o;
  }
}

// y = scale * (A[M,K] @ W[N,K]^T + bias).  One wave -> 32x64 tile
// (2 M-subtiles x 4 N-subtiles = 8 WMMAs per 32-wide k-step, B reused 2x).
// MODE 0: bf16 out in [B,H,S,hd]   (Q, K)
// MODE 1: bf16 out in [B,H,hd,S]   (V transposed)
// MODE 2: f32  out in [M,N]        (final projection)
template <int MODE>
__global__ void __launch_bounds__(256)
gemm_wmma_kernel(const bf16_t* __restrict__ A, const bf16_t* __restrict__ W,
                 const float* __restrict__ bias, void* __restrict__ Out,
                 int M, int N, int K, float scale) {
  const int lane = threadIdx.x & 31;
  const int wid  = blockIdx.x * (blockDim.x >> 5) + (threadIdx.x >> 5);
  const int nt4  = N >> 6;
  const int mt   = wid / nt4;            // 32-row tile index
  const int nb   = (wid % nt4) << 6;
  if (mt >= (M >> 5)) return;            // wave-uniform, EXEC stays all-ones
  const int lm   = lane & 15;
  const int half = lane >> 4;

  const bf16_t* ap0 = A + (size_t)(mt * 32 + lm) * K + half * 16;
  const bf16_t* ap1 = ap0 + (size_t)16 * K;
  const bf16_t* wp  = W + (size_t)(nb + lm) * K + half * 16;

  // software-pipelined K loop: fragments for step k loaded during step k-32
  v16bf a0 = load_a_frag(ap0);
  v16bf a1 = load_a_frag(ap1);
  v16bf b0 = load_b_frag(wp);
  v16bf b1 = load_b_frag(wp + (size_t)16 * K);
  v16bf b2 = load_b_frag(wp + (size_t)32 * K);
  v16bf b3 = load_b_frag(wp + (size_t)48 * K);

  v8f c00 = {}, c01 = {}, c02 = {}, c03 = {};
  v8f c10 = {}, c11 = {}, c12 = {}, c13 = {};
  for (int k0 = 32; k0 < K; k0 += 32) {
    v16bf an0 = load_a_frag(ap0 + k0);
    v16bf an1 = load_a_frag(ap1 + k0);
    v16bf bn0 = load_b_frag(wp + k0);
    v16bf bn1 = load_b_frag(wp + (size_t)16 * K + k0);
    v16bf bn2 = load_b_frag(wp + (size_t)32 * K + k0);
    v16bf bn3 = load_b_frag(wp + (size_t)48 * K + k0);
    c00 = wmma_bf16(a0, b0, c00);
    c01 = wmma_bf16(a0, b1, c01);
    c02 = wmma_bf16(a0, b2, c02);
    c03 = wmma_bf16(a0, b3, c03);
    c10 = wmma_bf16(a1, b0, c10);
    c11 = wmma_bf16(a1, b1, c11);
    c12 = wmma_bf16(a1, b2, c12);
    c13 = wmma_bf16(a1, b3, c13);
    a0 = an0; a1 = an1; b0 = bn0; b1 = bn1; b2 = bn2; b3 = bn3;
  }
  c00 = wmma_bf16(a0, b0, c00);
  c01 = wmma_bf16(a0, b1, c01);
  c02 = wmma_bf16(a0, b2, c02);
  c03 = wmma_bf16(a0, b3, c03);
  c10 = wmma_bf16(a1, b0, c10);
  c11 = wmma_bf16(a1, b1, c11);
  c12 = wmma_bf16(a1, b2, c12);
  c13 = wmma_bf16(a1, b3, c13);

  v8f accs[2][4] = {{c00, c01, c02, c03}, {c10, c11, c12, c13}};
#pragma unroll
  for (int i = 0; i < 2; ++i) {
    const int rowbase = mt * 32 + i * 16 + half * 8;  // C: lane>=16 -> rows 8..15
#pragma unroll
    for (int j = 0; j < 4; ++j) {
      const int col = nb + j * 16 + lm;
      const float bv = bias[col];
#pragma unroll
      for (int r = 0; r < 8; ++r) {
        const int row = rowbase + r;
        float y = (accs[i][j][r] + bv) * scale;
        if (MODE == 2) {
          ((float*)Out)[(size_t)row * N + col] = y;
        } else {
          const int b = row >> 11, s = row & (SDIM - 1);
          const int h = col >> 6,  d = col & (HDIM - 1);
          bf16_t* o = (bf16_t*)Out;
          if (MODE == 0)
            o[(((size_t)(b * NHEADS + h)) * SDIM + s) * HDIM + d] = (bf16_t)y;
          else
            o[(((size_t)(b * NHEADS + h)) * HDIM + d) * SDIM + s] = (bf16_t)y;
        }
      }
    }
  }
}

struct KFrag { v16bf f00, f01, f10, f11; };
static __device__ __forceinline__ KFrag
load_kfrag(const bf16_t* Kb, int nbase, int lm, int half) {
  const bf16_t* kp0 = Kb + (size_t)(nbase + lm) * HDIM + half * 16;
  const bf16_t* kp1 = Kb + (size_t)(nbase + 16 + lm) * HDIM + half * 16;
  KFrag k;
  k.f00 = load_a_frag(kp0);
  k.f01 = load_a_frag(kp0 + 32);
  k.f10 = load_a_frag(kp1);
  k.f11 = load_a_frag(kp1 + 32);
  return k;
}

// Causal flash attention. grid = (S/64, B*H), block = 128 (4 waves).
// One wave owns 16 query rows. Computes S^T = K.Q^T and O^T = V^T.P^T so each
// lane owns one query row -> per-lane softmax stats, scalar O rescale.
// Q is pre-scaled by 1/sqrt(hd) in the projection.
__global__ void __launch_bounds__(128)
attn_kernel(const bf16_t* __restrict__ Q, const bf16_t* __restrict__ Kc,
            const bf16_t* __restrict__ Vt, bf16_t* __restrict__ O) {
  const int lane = threadIdx.x & 31;
  const int wave = threadIdx.x >> 5;
  const int lm   = lane & 15;
  const int half = lane >> 4;
  const int bh = blockIdx.y;
  const int b  = bh >> 4;
  const int h  = bh & (NHEADS - 1);
  const int m0 = (blockIdx.x * 4 + wave) * 16;
  const int q  = m0 + lm;             // query row owned by this lane

  const bf16_t* Qb = Q  + (size_t)bh * SDIM * HDIM;
  const bf16_t* Kb = Kc + (size_t)bh * SDIM * HDIM;
  const bf16_t* Vb = Vt + (size_t)bh * HDIM * SDIM;

  // Q as B-fragment (lane = column m, contiguous d per half)
  const v16bf qf0 = load_b_frag(Qb + (size_t)q * HDIM + half * 16);
  const v16bf qf1 = load_b_frag(Qb + (size_t)q * HDIM + 32 + half * 16);

  v8f ot0 = {}, ot1 = {}, ot2 = {}, ot3 = {};
  float m_run = -1e30f, l_run = 0.f;

  const int kv_end = m0 + 16;         // causal bound (exclusive)
  const int nmask  = m0 & ~31;        // the single step that needs masking

  KFrag kf = load_kfrag(Kb, 0, lm, half);   // prefetched fragments for step 0

  for (int nbase = 0; nbase < kv_end; nbase += 32) {
    // ---- issue V loads early (consumed after the softmax VALU work)
    const bf16_t* vp = Vb + (size_t)lm * SDIM + nbase + half * 16;
    v16bf vf0 = load_a_frag(vp);
    v16bf vf1 = load_a_frag(vp + (size_t)16 * SDIM);
    v16bf vf2 = load_a_frag(vp + (size_t)32 * SDIM);
    v16bf vf3 = load_a_frag(vp + (size_t)48 * SDIM);

    // ---- scores: two 16x16 tiles of S^T = K.Q^T (K fragments prefetched)
    v8f st0 = {}, st1 = {};
    st0 = wmma_bf16(kf.f00, qf0, st0);
    st0 = wmma_bf16(kf.f01, qf1, st0);
    st1 = wmma_bf16(kf.f10, qf0, st1);
    st1 = wmma_bf16(kf.f11, qf1, st1);

    // ---- prefetch next step's K fragments
    if (nbase + 32 < kv_end) kf = load_kfrag(Kb, nbase + 32, lm, half);

    // ---- causal mask (wave-uniform: only the diagonal step) + per-lane max
    float s0[8], s1[8];
#pragma unroll
    for (int r = 0; r < 8; ++r) { s0[r] = st0[r]; s1[r] = st1[r]; }
    if (nbase == nmask) {
#pragma unroll
      for (int r = 0; r < 8; ++r) {
        const int n0 = nbase + half * 8 + r;
        if (n0 > q)      s0[r] = -1e30f;
        if (n0 + 16 > q) s1[r] = -1e30f;
      }
    }
    float lmax = -1e30f;
#pragma unroll
    for (int r = 0; r < 8; ++r) lmax = fmaxf(lmax, fmaxf(s0[r], s1[r]));
    lmax = fmaxf(lmax, __shfl_xor(lmax, 16, 32));   // combine n-halves of row q
    const float m_new = fmaxf(m_run, lmax);
    const float corr  = __expf(m_run - m_new);
    m_run = m_new;

    float p0[8], p1[8], lsum = 0.f;
#pragma unroll
    for (int r = 0; r < 8; ++r) {
      p0[r] = __expf(s0[r] - m_new);
      p1[r] = __expf(s1[r] - m_new);
      lsum += p0[r] + p1[r];
    }
    l_run = l_run * corr + lsum;

    // ---- repack P^T into a B-fragment: lane needs n = half*16 + {0..15};
    // exchange interleaved halves between lane pairs (L, L^16).
    union { v16bf v; bf16_t e[16]; } pf;
#pragma unroll
    for (int r = 0; r < 8; ++r) {
      const float send = half ? p0[r] : p1[r];
      const float recv = __shfl_xor(send, 16, 32);
      const float e_lo = half ? recv : p0[r];   // n = half*16 + r
      const float e_hi = half ? p1[r] : recv;   // n = half*16 + 8 + r
      pf.e[r]     = (bf16_t)e_lo;
      pf.e[8 + r] = (bf16_t)e_hi;
    }

    // ---- rescale running O^T (per-lane scalar) and accumulate V^T . P^T
    ot0 *= corr; ot1 *= corr; ot2 *= corr; ot3 *= corr;
    ot0 = wmma_bf16(vf0, pf.v, ot0);
    ot1 = wmma_bf16(vf1, pf.v, ot1);
    ot2 = wmma_bf16(vf2, pf.v, ot2);
    ot3 = wmma_bf16(vf3, pf.v, ot3);
  }

  const float l_tot = l_run + __shfl_xor(l_run, 16, 32);
  const float inv = 1.f / l_tot;
  // O^T C-layout: lane = column (= its query row q); VGPR r -> d = j*16+half*8+r
  bf16_t* Ob = O + ((size_t)b * SDIM + q) * DMODEL + h * HDIM + half * 8;
  v8f ots[4] = {ot0, ot1, ot2, ot3};
#pragma unroll
  for (int j = 0; j < 4; ++j) {
    v8bf pack;
#pragma unroll
    for (int r = 0; r < 8; ++r) pack[r] = (bf16_t)(ots[j][r] * inv);
    *(v8bf*)(Ob + j * 16) = pack;
  }
}

extern "C" void kernel_launch(void* const* d_in, const int* in_sizes, int n_in,
                              void* d_out, int out_size, void* d_ws, size_t ws_size,
                              hipStream_t stream) {
  const float* x  = (const float*)d_in[0];
  const float* Wq = (const float*)d_in[1];
  const float* bq = (const float*)d_in[2];
  const float* Wk = (const float*)d_in[3];
  const float* bk = (const float*)d_in[4];
  const float* Wv = (const float*)d_in[5];
  const float* bv = (const float*)d_in[6];
  const float* Wo = (const float*)d_in[7];
  const float* bo = (const float*)d_in[8];

  char* ws = (char*)d_ws;
  const size_t MB = 1024ull * 1024ull;
  bf16_t* xb  = (bf16_t*)(ws);             //  8 MB  [4096,1024] bf16
  bf16_t* Wqb = (bf16_t*)(ws + 8  * MB);   //  2 MB
  bf16_t* Wkb = (bf16_t*)(ws + 10 * MB);   //  2 MB
  bf16_t* Wvb = (bf16_t*)(ws + 12 * MB);   //  2 MB
  bf16_t* Wob = (bf16_t*)(ws + 14 * MB);   //  2 MB
  bf16_t* Qb  = (bf16_t*)(ws + 16 * MB);   //  8 MB  [B,H,S,hd], pre-scaled
  bf16_t* Kb  = (bf16_t*)(ws + 24 * MB);   //  8 MB  [B,H,S,hd]
  bf16_t* Vtb = (bf16_t*)(ws + 32 * MB);   //  8 MB  [B,H,hd,S]
  bf16_t* Ob  = (bf16_t*)(ws + 40 * MB);   //  8 MB  [B,S,D]

  auto cvt = [&](const float* src, bf16_t* dst, int n) {
    int n4 = n / 4;
    cvt_f32_to_bf16<<<(n4 + 255) / 256, 256, 0, stream>>>(src, dst, n4);
  };
  cvt(x,  xb,  MROWS * DMODEL);
  cvt(Wq, Wqb, DMODEL * DMODEL);
  cvt(Wk, Wkb, DMODEL * DMODEL);
  cvt(Wv, Wvb, DMODEL * DMODEL);
  cvt(Wo, Wob, DMODEL * DMODEL);

  const int nwaves = (MROWS / 32) * (DMODEL / 64);   // 2048
  const int blocks = nwaves / 8;                     // 8 waves / 256-thread block
  gemm_wmma_kernel<0><<<blocks, 256, 0, stream>>>(xb, Wqb, bq, Qb,  MROWS, DMODEL, DMODEL, SCALE_F);
  gemm_wmma_kernel<0><<<blocks, 256, 0, stream>>>(xb, Wkb, bk, Kb,  MROWS, DMODEL, DMODEL, 1.0f);
  gemm_wmma_kernel<1><<<blocks, 256, 0, stream>>>(xb, Wvb, bv, Vtb, MROWS, DMODEL, DMODEL, 1.0f);

  attn_kernel<<<dim3(SDIM / 64, BATCH * NHEADS), 128, 0, stream>>>(Qb, Kb, Vtb, Ob);

  gemm_wmma_kernel<2><<<blocks, 256, 0, stream>>>(Ob, Wob, bo, d_out, MROWS, DMODEL, DMODEL, 1.0f);
}
